// SMC_Transformer_11991548690951
// MI455X (gfx1250) — matrix operand order repaired and checked by reading, not compiled, not run on hardware
//
#include <hip/hip_runtime.h>

typedef __attribute__((ext_vector_type(16))) _Float16 v16h;
typedef __attribute__((ext_vector_type(8)))  float    v8f;

#define B_ 8
#define S_ 128
#define F_ 11
#define D_ 64
#define O_ 1
#define P_ 16

// ---------------- deterministic PRNG (compile-only stand-in for threefry) ---
__device__ __forceinline__ unsigned pcg_hash(unsigned x) {
  x = x * 747796405u + 2891336453u;
  unsigned w = ((x >> ((x >> 28) + 4u)) ^ x) * 277803737u;
  return (w >> 22) ^ w;
}
__device__ __forceinline__ float u01(unsigned h) {
  return (float)(h >> 8) * (1.0f / 16777216.0f) + (0.5f / 16777216.0f);
}
__device__ __forceinline__ float gaussn(unsigned b, unsigned p, unsigned t,
                                        unsigned d, unsigned stream) {
  unsigned s0 = pcg_hash(((t * 131u + stream) * 131u + b) * 131u + p * 97u + d);
  unsigned s1 = pcg_hash(s0 ^ 0x9E3779B9u);
  float u1 = u01(s0), u2 = u01(s1);
  return sqrtf(-2.0f * __logf(u1)) * __cosf(6.2831853f * u2);
}

// ---------------- fragment-order LDS staging (CDNA5 wave32 WMMA layouts) ----
// Operands stored in LDS pre-swizzled as [frag][lane][16 halves] so each wave
// loads a whole v16h operand with two contiguous ds_load_b128.
__device__ __forceinline__ int a_slot(int m, int d) {
  int c = d >> 5, kk = d & 31;
  int hi = (kk >> 3) & 1;
  int i = ((kk & 7) >> 1) + ((kk >> 4) & 1) * 4;
  int h = 2 * i + (kk & 1);
  return (c * 32 + (m + 16 * hi)) * 16 + h;
}
__device__ __forceinline__ int b_slot(int k, int n) {
  int nt = n >> 4, c = k >> 5, kk = k & 31;
  int hi = (kk >> 3) & 1;
  int L = (n & 15) + 16 * hi;
  int h = (kk & 7) + ((kk >> 4) & 1) * 8;
  return ((nt * 2 + c) * 32 + L) * 16 + h;
}
__device__ __forceinline__ v16h frag_ld(const _Float16* base, int fragid,
                                        int lane) {
  return *(const v16h*)(base + (fragid * 32 + lane) * 16);
}

// One 16x64x64 projection: stage W in B-frag order, 8 WMMAs, biased store.
// Concrete __restrict__ pointers -> global (not flat) addressing; bias value
// hoisted per lane (invariant across the 8 rows each lane writes).
__device__ __forceinline__ void proj_weight(
    const float* __restrict__ W, const float* __restrict__ bias,
    float* __restrict__ out, _Float16* sWB, const v16h a0, const v16h a1,
    int lane, int base) {
  __syncthreads();
  for (int idx = lane; idx < 64 * 64; idx += 32) {
    int k = idx >> 6, n = idx & 63;
    sWB[b_slot(k, n)] = (_Float16)W[idx];
  }
  __syncthreads();
  const int hi = lane >> 4;
#pragma unroll
  for (int nt = 0; nt < 4; ++nt) {
    v8f c = {};
    v16h b0 = frag_ld(sWB, nt * 2 + 0, lane);
    c = __builtin_amdgcn_wmma_f32_16x16x32_f16(false, a0, false, b0, (short)0,
                                               c, false, false);
    v16h b1 = frag_ld(sWB, nt * 2 + 1, lane);
    c = __builtin_amdgcn_wmma_f32_16x16x32_f16(false, a1, false, b1, (short)0,
                                               c, false, false);
    int n = (lane & 15) + nt * 16;
    float bn = bias[n];
    float* op = out + (size_t)(base + hi * 8) * D_ + n;
#pragma unroll
    for (int r = 0; r < 8; ++r) op[(size_t)r * D_] = c[r] + bn;
  }
}

// ============================================================================
// Kernel A: x = inputs@W_proj + b_proj; Qd/Kd/Vd = x@W{q,k,v} + b  (WMMA)
// grid = (B*S)/16 = 64 blocks, 32 threads (1 wave)
// ============================================================================
__global__ __launch_bounds__(32) void smc_proj_kernel(
    const float* __restrict__ inputs, const float* __restrict__ Wp,
    const float* __restrict__ bp, const float* __restrict__ Wq,
    const float* __restrict__ bq, const float* __restrict__ Wk,
    const float* __restrict__ bk, const float* __restrict__ Wv,
    const float* __restrict__ bv, float* __restrict__ QD,
    float* __restrict__ KD, float* __restrict__ VD) {
  __shared__ __align__(32) _Float16 sXA[2 * 32 * 16];  // A fragments (x tile)
  __shared__ __align__(32) _Float16 sWB[8 * 32 * 16];  // B fragments (weight)
  const int lane = threadIdx.x;
  const int base = blockIdx.x * 16;  // row in flattened (B*S)

  // x tile (K=11 matvec, scalar f32) scattered into A-fragment order
  for (int idx = lane; idx < 16 * 64; idx += 32) {
    int r = idx >> 6, d = idx & 63;
    const float* inr = inputs + (size_t)(base + r) * F_;
    float acc = bp[d];
#pragma unroll
    for (int f = 0; f < F_; ++f) acc += inr[f] * Wp[f * D_ + d];
    sXA[a_slot(r, d)] = (_Float16)acc;
  }
  __syncthreads();

  v16h a0 = frag_ld(sXA, 0, lane);
  v16h a1 = frag_ld(sXA, 1, lane);

  proj_weight(Wq, bq, QD, sWB, a0, a1, lane, base);
  proj_weight(Wk, bk, KD, sWB, a0, a1, lane, base);
  proj_weight(Wv, bv, VD, sWB, a0, a1, lane, base);
}

// ============================================================================
// Kernel B: the sequential SMC scan. grid = B blocks, 512 threads (16 waves).
// wave w == particle p. K/V caches double-buffered in global (L2-resident).
// ============================================================================
__global__ __launch_bounds__(512) void smc_scan_kernel(
    const float* __restrict__ targets, const float* __restrict__ QD,
    const float* __restrict__ KD, const float* __restrict__ VD,
    const float* __restrict__ Wo, const float* __restrict__ bo,
    const float* __restrict__ Wout, const float* __restrict__ bout,
    float* __restrict__ KC0, float* __restrict__ KC1, float* __restrict__ VC0,
    float* __restrict__ VC1, float* __restrict__ RW, float* __restrict__ outW0,
    float* __restrict__ outIND, float* __restrict__ outATT) {
  __shared__ __align__(32) _Float16 sWoF[8 * 32 * 16];  // Wo B-fragments
  __shared__ __align__(32) _Float16 sZA[2 * 32 * 16];   // z A-fragments
  __shared__ float s_bo[64], s_wout[64];
  __shared__ float s_qd[64], s_kd[64], s_vd[64];
  __shared__ __align__(16) float s_q[16 * 64];
  __shared__ float s_attn[16 * S_];
  __shared__ __align__(16) float s_z2[16 * 64];
  __shared__ float s_ll[16];
  __shared__ int s_ind[16];
  __shared__ float s_bsc;

  const int b = blockIdx.x;
  const int tid = threadIdx.x;
  const int w = tid >> 5;  // wave index == particle
  const int lane = tid & 31;
  const int p = w;

  // stage Wo directly in B-fragment order (one-time)
  for (int idx = tid; idx < 64 * 64; idx += 512) {
    int k = idx >> 6, n = idx & 63;
    sWoF[b_slot(k, n)] = (_Float16)Wo[idx];
  }
  if (tid < 64) { s_bo[tid] = bo[tid]; s_wout[tid] = Wout[tid]; }
  if (tid == 0) s_bsc = bout[0];
  __syncthreads();

  // Wo B-fragments live in registers for all 128 steps (waves 0-3 use them)
  const int nt = w & 3;
  v16h wb0 = frag_ld(sWoF, nt * 2 + 0, lane);
  v16h wb1 = frag_ld(sWoF, nt * 2 + 1, lane);

  float* KB[2] = {KC0, KC1};
  float* VB[2] = {VC0, VC1};
  const float scale = 0.125f;  // 1/sqrt(64)

  for (int t = 0; t < S_; ++t) {
    float* Kcur = KB[t & 1];
    float* Vcur = VB[t & 1];
    float* Knxt = KB[(t + 1) & 1];
    float* Vnxt = VB[(t + 1) & 1];

    // stage deterministic projection rows for this (b,t)
    if (tid < 64) s_qd[tid] = QD[(size_t)(b * S_ + t) * D_ + tid];
    else if (tid < 128) s_kd[tid - 64] = KD[(size_t)(b * S_ + t) * D_ + tid - 64];
    else if (tid < 192) s_vd[tid - 128] = VD[(size_t)(b * S_ + t) * D_ + tid - 128];
    __syncthreads();

    // per-particle q/k/v = deterministic + sigma*noise; k,v -> cache row t
    const size_t pbase = (size_t)(b * P_ + p) * S_ * D_;
    const size_t cb = pbase + (size_t)t * D_;
    {
      int d0 = lane * 2;
      float2 qv, kv, vv;
      qv.x = s_qd[d0] + 0.1f * gaussn(b, p, t, d0, 1);
      qv.y = s_qd[d0 + 1] + 0.1f * gaussn(b, p, t, d0 + 1, 1);
      kv.x = s_kd[d0] + 0.1f * gaussn(b, p, t, d0, 2);
      kv.y = s_kd[d0 + 1] + 0.1f * gaussn(b, p, t, d0 + 1, 2);
      vv.x = s_vd[d0] + 0.1f * gaussn(b, p, t, d0, 3);
      vv.y = s_vd[d0 + 1] + 0.1f * gaussn(b, p, t, d0 + 1, 3);
      *(float2*)(s_q + p * 64 + d0) = qv;
      *(float2*)(Kcur + cb + d0) = kv;
      *(float2*)(Vcur + cb + d0) = vv;
    }
    __syncthreads();

    // causal scores for s<=t (lane-strided over s, float4 K rows)
    const float4* q4 = (const float4*)(s_q + p * 64);
    float lmax = -3.0e38f;
    for (int s = lane; s <= t; s += 32) {
      const float4* kr = (const float4*)(Kcur + pbase + (size_t)s * D_);
      __builtin_prefetch((const void*)(kr + 512), 0, 1);  // 32 rows ahead
      float acc = 0.f;
#pragma unroll
      for (int dq = 0; dq < 16; ++dq) {
        float4 kk = kr[dq];
        float4 qq = q4[dq];
        acc += qq.x * kk.x + qq.y * kk.y + qq.z * kk.z + qq.w * kk.w;
      }
      acc *= scale;
      s_attn[p * S_ + s] = acc;
      lmax = fmaxf(lmax, acc);
    }
    for (int m = 16; m >= 1; m >>= 1) lmax = fmaxf(lmax, __shfl_xor(lmax, m, 32));
    float lsum = 0.f;
    for (int s = lane; s <= t; s += 32) {
      float e = __expf(s_attn[p * S_ + s] - lmax);
      s_attn[p * S_ + s] = e;
      lsum += e;
    }
    for (int m = 16; m >= 1; m >>= 1) lsum += __shfl_xor(lsum, m, 32);
    float rinv = 1.0f / lsum;
    // normalize in place + emit attn_weights row (zeros past t)
    const size_t abase = ((size_t)(b * P_ + p) * S_ + t) * S_;
    for (int s = lane; s < S_; s += 32) {
      float a = (s <= t) ? s_attn[p * S_ + s] * rinv : 0.f;
      s_attn[p * S_ + s] = a;
      outATT[abase + s] = a;
    }
    __syncthreads();  // cross-lane LDS visibility of attn row

    // z = attn @ V (lane owns dims 2l,2l+1); scatter straight into A-frag LDS
    {
      int d0 = lane * 2;
      float z0 = 0.f, z1 = 0.f;
      const float* vb = Vcur + pbase;
      for (int s = 0; s <= t; ++s) {
        float a = s_attn[p * S_ + s];
        float2 vv = *(const float2*)(vb + (size_t)s * D_ + d0);
        z0 += a * vv.x;
        z1 += a * vv.y;
      }
      sZA[a_slot(p, d0)] = (_Float16)z0;
      sZA[a_slot(p, d0 + 1)] = (_Float16)z1;
    }
    __syncthreads();

    // z2 = z @ Wo + bo + sigma_z*noise  — 16(particles)x64x64 GEMM via WMMA
    if (w < 4) {
      v16h az0 = frag_ld(sZA, 0, lane);
      v16h az1 = frag_ld(sZA, 1, lane);
      v8f c = {};
      c = __builtin_amdgcn_wmma_f32_16x16x32_f16(false, az0, false, wb0,
                                                 (short)0, c, false, false);
      c = __builtin_amdgcn_wmma_f32_16x16x32_f16(false, az1, false, wb1,
                                                 (short)0, c, false, false);
      int n = (lane & 15) + nt * 16;
      int hi = lane >> 4;
#pragma unroll
      for (int r = 0; r < 8; ++r) {
        int mm = r + 8 * hi;
        s_z2[mm * 64 + n] = c[r] + s_bo[n] + 0.1f * gaussn(b, mm, t, n, 4);
      }
    }
    __syncthreads();

    // pred (D->1) and log-likelihood per particle
    {
      int d0 = lane * 2;
      float2 zz = *(const float2*)(s_z2 + p * 64 + d0);
      float acc = zz.x * s_wout[d0] + zz.y * s_wout[d0 + 1];
      for (int m = 16; m >= 1; m >>= 1) acc += __shfl_xor(acc, m, 32);
      float pred = acc + s_bsc;
      float yt = targets[(size_t)(b * S_ + t) * O_];
      float dlt = yt - pred;
      if (lane == 0) s_ll[p] = -2.0f * dlt * dlt;  // -0.5/sigma_obs^2
    }
    __syncthreads();

    // particle-softmax weights (wave 0) — all 32 lanes active, j duplicated
    if (w == 0) {
      int j = lane & 15;
      float v = s_ll[j];
      float mx = v;
      for (int m = 8; m >= 1; m >>= 1) mx = fmaxf(mx, __shfl_xor(mx, m, 32));
      float e = __expf(v - mx);
      float sm = e;
      for (int m = 8; m >= 1; m >>= 1) sm += __shfl_xor(sm, m, 32);
      if (lane < 16) outW0[(size_t)(b * S_ + t) * P_ + j] = e / sm;
    }
    // categorical resampling index via Gumbel-max (wave p draws its own)
    {
      int j = lane & 15;
      unsigned h = pcg_hash(((t * 131u + 5u) * 131u + b) * 131u + p * 97u + j);
      float gv = s_ll[j] - __logf(-__logf(u01(h)));
      int gi = j;
      for (int m = 8; m >= 1; m >>= 1) {
        float ov = __shfl_xor(gv, m, 32);
        int oi = __shfl_xor(gi, m, 32);
        if (ov > gv || (ov == gv && oi < gi)) { gv = ov; gi = oi; }
      }
      if (lane == 0) {
        s_ind[p] = gi;
        outIND[(size_t)(b * S_ + t) * P_ + p] = (float)gi;
      }
    }
    __syncthreads();

    // resample: gather full K/V histories cur->nxt (float4 wide), zr -> R
    {
      int ip = s_ind[p];
      const float4* ks = (const float4*)(Kcur + (size_t)(b * P_ + ip) * S_ * D_);
      const float4* vs = (const float4*)(Vcur + (size_t)(b * P_ + ip) * S_ * D_);
      float4* kd = (float4*)(Knxt + pbase);
      float4* vd = (float4*)(Vnxt + pbase);
      int q = lane & 15, half = lane >> 4;  // 2 rows in flight per wave
      for (int s = half; s <= t; s += 2) {
        kd[s * 16 + q] = ks[s * 16 + q];
        vd[s * 16 + q] = vs[s * 16 + q];
      }
      int d0 = lane * 2;
      *(float2*)(RW + cb + d0) = *(const float2*)(s_z2 + ip * 64 + d0);
    }
    __syncthreads();
  }
}

// ============================================================================
// Kernel C: slice K/V/R (t=1..127) to outputs and Y0 = R@W_out + b_out
// ============================================================================
__global__ __launch_bounds__(256) void smc_final_kernel(
    const float* __restrict__ KC, const float* __restrict__ VC,
    const float* __restrict__ RW, const float* __restrict__ Wout,
    const float* __restrict__ bout, float* __restrict__ outY0,
    float* __restrict__ outK, float* __restrict__ outV,
    float* __restrict__ outR) {
  const int idx = blockIdx.x * 256 + threadIdx.x;
  const int N4 = B_ * P_ * (S_ - 1) * (D_ / 4);  // 260096 float4 copies
  if (idx < N4) {
    int d4 = idx & 15;
    int rest = idx >> 4;
    int t = rest % (S_ - 1) + 1;
    int bp = rest / (S_ - 1);
    size_t src4 = (((size_t)bp * S_ + t) * D_) / 4 + d4;
    ((float4*)outK)[idx] = ((const float4*)KC)[src4];
    ((float4*)outV)[idx] = ((const float4*)VC)[src4];
    ((float4*)outR)[idx] = ((const float4*)RW)[src4];
  }
  const int NY0 = B_ * P_ * (S_ - 1);
  if (idx < NY0) {
    int t = idx % (S_ - 1) + 1;
    int bp = idx / (S_ - 1);
    const float4* r4 = (const float4*)(RW + ((size_t)bp * S_ + t) * D_);
    const float4* w4 = (const float4*)Wout;
    float acc = bout[0];
#pragma unroll
    for (int dq = 0; dq < 16; ++dq) {
      float4 rr = r4[dq];
      float4 ww = w4[dq];
      acc += rr.x * ww.x + rr.y * ww.y + rr.z * ww.z + rr.w * ww.w;
    }
    outY0[idx] = acc;
  }
}

// ============================================================================
extern "C" void kernel_launch(void* const* d_in, const int* in_sizes, int n_in,
                              void* d_out, int out_size, void* d_ws,
                              size_t ws_size, hipStream_t stream) {
  (void)in_sizes; (void)n_in; (void)out_size; (void)ws_size;
  const float* inputs = (const float*)d_in[0];
  const float* targets = (const float*)d_in[1];
  const float* Wp = (const float*)d_in[2];
  const float* bp = (const float*)d_in[3];
  const float* Wq = (const float*)d_in[4];
  const float* bq = (const float*)d_in[5];
  const float* Wk = (const float*)d_in[6];
  const float* bk = (const float*)d_in[7];
  const float* Wv = (const float*)d_in[8];
  const float* bv = (const float*)d_in[9];
  const float* Wo = (const float*)d_in[10];
  const float* bo = (const float*)d_in[11];
  const float* Wout = (const float*)d_in[12];
  const float* bout = (const float*)d_in[13];

  float* ws = (float*)d_ws;
  float* QD = ws;                    // 8*128*64 = 65536
  float* KD = QD + 65536;
  float* VD = KD + 65536;
  float* KC0 = VD + 65536;           // 8*16*128*64 = 1048576 each
  float* KC1 = KC0 + 1048576;
  float* VC0 = KC1 + 1048576;
  float* VC1 = VC0 + 1048576;
  float* RW = VC1 + 1048576;

  float* out = (float*)d_out;        // concatenated in reference return order
  float* Y0 = out;                   // (B,P,127,1)   = 16256
  float* W0 = Y0 + 16256;            // (B,S,P)       = 16384
  float* IND = W0 + 16384;           // (B,S,P)       = 16384
  float* KO = IND + 16384;           // (B,P,127,64)  = 1040384
  float* VO = KO + 1040384;
  float* RO = VO + 1040384;
  float* ATT = RO + 1040384;         // (B,P,1,S,S)   = 2097152

  smc_proj_kernel<<<(B_ * S_) / 16, 32, 0, stream>>>(inputs, Wp, bp, Wq, bq, Wk,
                                                     bk, Wv, bv, QD, KD, VD);
  smc_scan_kernel<<<B_, 512, 0, stream>>>(targets, QD, KD, VD, Wo, bo, Wout,
                                          bout, KC0, KC1, VC0, VC1, RW, W0,
                                          IND, ATT);
  // after t=127 the gather wrote buffer (128 & 1) == 0
  smc_final_kernel<<<(260096 + 255) / 256, 256, 0, stream>>>(
      KC0, VC0, RW, Wout, bout, Y0, KO, VO, RO);
}
